// RSSMCore_30657476558860
// MI455X (gfx1250) — matrix operand order repaired and checked by reading, not compiled, not run on hardware
//
#include <hip/hip_runtime.h>

typedef __attribute__((ext_vector_type(16))) __bf16 v16bf;
typedef __attribute__((ext_vector_type(8)))  __bf16 v8bf;
typedef __attribute__((ext_vector_type(8)))  float  v8f;

#define TT 256
#define BB 1024

// ---- workspace layout (bf16 elements), all offsets multiples of 16 ----
constexpr size_t O_ZA  = 0;                       // [208][64]  : [z_w | a_w]
constexpr size_t O_RZ  = O_ZA  + (size_t)208*64;  // [416][448] : r,z gates of [W_ih | W_hh]
constexpr size_t O_IN  = O_RZ  + (size_t)416*448; // [208][224] : n gate of W_ih
constexpr size_t O_HN  = O_IN  + (size_t)208*224; // [208][224] : n gate of W_hh
constexpr size_t O_PC  = O_HN  + (size_t)208*224; // [208][480] : [post_h_w | post_e_w]
constexpr size_t O_PO  = O_PC  + (size_t)208*480; // [64][224]  : post_w
constexpr size_t O_PRH = O_PO  + (size_t)64*224;  // [208][224] : prior_h_w
constexpr size_t O_PR  = O_PRH + (size_t)208*224; // [64][224]  : prior_w

// ---------------- fast math (hardware v_exp/v_log/v_rcp, branchless) ----------------
__device__ __forceinline__ float eluf(float x) {
    float e = __expf(x) - 1.f;          // discarded when x>0; inf harmless
    return x > 0.f ? x : e;
}
__device__ __forceinline__ float sigmf(float x) {
    return __builtin_amdgcn_rcpf(1.f + __expf(-x));
}
__device__ __forceinline__ float tanh_fast(float x) {
    float cx = fminf(15.f, fmaxf(-15.f, x));
    float e2 = __expf(2.f * cx);
    return (e2 - 1.f) * __builtin_amdgcn_rcpf(e2 + 1.f);
}
__device__ __forceinline__ float softplusf(float x) {
    float e  = __expf(fminf(x, 20.f));
    float sp = __logf(1.f + e);
    return x > 20.f ? x : sp;
}

// ---------------- fragment loaders ----------------
// A: row-major bf16 in LDS, row stride lda, 32-wide k window at kb.
__device__ __forceinline__ v16bf load_a(const __bf16* A, int lda, int kb, int lane)
{
    const __bf16* ap = A + (lane & 15) * lda + kb + 16 * (lane >> 4);
    v8bf a0 = *(const v8bf*)(ap);
    v8bf a1 = *(const v8bf*)(ap + 8);
    return __builtin_shufflevector(a0, a1, 0,1,2,3,4,5,6,7,8,9,10,11,12,13,14,15);
}
// B: weight rows [out=N][in=K] row-major bf16; Wrow0 points at first row of this N-tile.
__device__ __forceinline__ v16bf load_b(const __bf16* Wrow0, int ldb, int kb, int lane)
{
    return *(const v16bf*)(Wrow0 + (size_t)(lane & 15) * ldb + kb + 16 * (lane >> 4));
}
__device__ __forceinline__ v8f wmma_bf16(v16bf a, v16bf b, v8f c)
{
    return __builtin_amdgcn_wmma_f32_16x16x32_bf16(false, a, false, b, (short)0, c, false, false);
}
// generic single-accumulator tile
__device__ __forceinline__ v8f wmma_tile(const __bf16* A, int lda, int ka0, int ksteps,
                                         const __bf16* Wrow0, int ldb, int lane)
{
    v8f acc = {0.f, 0.f, 0.f, 0.f, 0.f, 0.f, 0.f, 0.f};
    #pragma unroll
    for (int k = 0; k < ksteps; ++k) {
        v16bf a = load_a(A, lda, ka0 + 32 * k, lane);
        v16bf b = load_b(Wrow0, ldb, 32 * k, lane);
        acc = wmma_bf16(a, b, acc);
    }
    return acc;
}

// ---------------- weight prep: pad/convert [N][Ka]+[N][Kb] -> bf16 [Np][Kp] ----------------
__global__ void prep_weight(__bf16* __restrict__ dst,
                            const float* __restrict__ srcA, int Ka, int KpA,
                            const float* __restrict__ srcB, int Kb,
                            int N, int Np, int Kp)
{
    int i = blockIdx.x * 256 + threadIdx.x;
    if (i >= Np * Kp) return;
    int j = i / Kp, k = i - j * Kp;
    float v = 0.f;
    if (j < N) {
        if (k < KpA) { if (k < Ka) v = srcA[(size_t)j * Ka + k]; }
        else { int kk = k - KpA; if (kk < Kb) v = srcB[(size_t)j * Kb + kk]; }
    }
    dst[i] = (__bf16)v;
}

// ---------------- main scan kernel: one WG = 16 batch rows, persistent over T ----------------
__global__ void __launch_bounds__(256) rssm_scan(
    const float* __restrict__ embed, const float* __restrict__ action,
    const unsigned char* __restrict__ reset,
    const float* __restrict__ h0, const float* __restrict__ z0,
    const float* __restrict__ noise,
    const float* __restrict__ z_b, const float* __restrict__ gb_ih,
    const float* __restrict__ gb_hh,
    const float* __restrict__ poh_b, const float* __restrict__ po_b,
    const __bf16* __restrict__ Wza, const __bf16* __restrict__ Wrz,
    const __bf16* __restrict__ Win, const __bf16* __restrict__ Whn,
    const __bf16* __restrict__ Wpc, const __bf16* __restrict__ Wpo,
    float* __restrict__ posts, float* __restrict__ feat,
    float* __restrict__ lasth, float* __restrict__ lastz)
{
    __shared__ __attribute__((aligned(64))) __bf16 sAzh[16 * 448]; // [za(224) | h(224)]
    __shared__ __attribute__((aligned(64))) __bf16 sApo[16 * 480]; // [h_new(224) | e(256)]
    __shared__ __attribute__((aligned(64))) __bf16 sApi[16 * 224]; // za-input(64) / post_in(224)
    __shared__ float sH[16 * 200];
    __shared__ float sZ[16 * 30];
    __shared__ float sPost[16 * 64];
    __shared__ float sMask[16];

    const int tid  = threadIdx.x;
    const int lane = tid & 31;
    const int wid  = tid >> 5;
    const int hi   = lane >> 4;
    const int colL = lane & 15;
    const int b0   = blockIdx.x * 16;

    // init carry + zero pad columns (written once, never touched again)
    for (int i = tid; i < 16 * 200; i += 256) {
        int r = i / 200, k = i - r * 200;
        sH[i] = h0[(size_t)(b0 + r) * 200 + k];
    }
    for (int i = tid; i < 16 * 30; i += 256) {
        int r = i / 30, k = i - r * 30;
        sZ[i] = z0[(size_t)(b0 + r) * 30 + k];
    }
    for (int i = tid; i < 16 * 24; i += 256) {
        int r = i / 24, k = 200 + (i - r * 24);
        sAzh[r * 448 + k]       = (__bf16)0.f;  // za pad
        sAzh[r * 448 + 224 + k] = (__bf16)0.f;  // h pad
        sApo[r * 480 + k]       = (__bf16)0.f;  // h_new pad
        sApi[r * 224 + k]       = (__bf16)0.f;  // post_in pad
    }
    __syncthreads();

    for (int t = 0; t < TT; ++t) {
        const size_t rowbase = (size_t)t * BB + b0;

        // ---- stage0a: reset mask ----
        if (tid < 16) sMask[tid] = reset[rowbase + tid] ? 0.f : 1.f;
        __syncthreads();

        // ---- stage0b: build [z*m | a] input, masked h, embed ----
        for (int i = tid; i < 16 * 64; i += 256) {
            int r = i >> 6, k = i & 63;
            float v = 0.f;
            if (k < 30)      { v = sZ[r * 30 + k] * sMask[r]; }
            else if (k < 37) { v = action[(rowbase + r) * 7 + (k - 30)]; }
            sApi[r * 224 + k] = (__bf16)v;
        }
        for (int i = tid; i < 16 * 200; i += 256) {
            int r = i / 200;
            sAzh[r * 448 + 224 + (i - r * 200)] = (__bf16)(sH[i] * sMask[r]);
        }
        for (int i = tid; i < 16 * 256; i += 256) {
            int r = i >> 8, k = i & 255;
            sApo[r * 480 + 224 + k] = (__bf16)embed[(rowbase + r) * 256 + k];
        }
        __syncthreads();

        // ---- GEMM1: za = elu([z|a] @ Wza^T + z_b) -> sAzh[:,0:200] ----
        for (int nt = wid; nt < 13; nt += 8) {
            v8f acc = wmma_tile(sApi, 224, 0, 2, Wza + (size_t)nt * 16 * 64, 64, lane);
            int col = nt * 16 + colL;
            if (col < 200) {
                float b = z_b[col];
                #pragma unroll
                for (int j = 0; j < 8; ++j) {
                    int r = j + 8 * hi;
                    sAzh[r * 448 + col] = (__bf16)eluf(acc[j] + b);
                }
            }
        }
        __syncthreads();

        // ---- GEMM2: fused GRU gates: one A load per k-step feeds 3 WMMAs ----
        for (int nt = wid; nt < 13; nt += 8) {
            const __bf16* wr = Wrz + (size_t)(nt * 16) * 448;
            const __bf16* wz = Wrz + (size_t)(208 + nt * 16) * 448;
            const __bf16* wi = Win + (size_t)(nt * 16) * 224;
            const __bf16* wh = Whn + (size_t)(nt * 16) * 224;
            v8f ar  = {0.f,0.f,0.f,0.f,0.f,0.f,0.f,0.f};
            v8f az  = {0.f,0.f,0.f,0.f,0.f,0.f,0.f,0.f};
            v8f ain = {0.f,0.f,0.f,0.f,0.f,0.f,0.f,0.f};
            v8f ahn = {0.f,0.f,0.f,0.f,0.f,0.f,0.f,0.f};
            #pragma unroll
            for (int k = 0; k < 14; ++k) {
                v16bf a  = load_a(sAzh, 448, 32 * k, lane);   // [za | h] sweep
                v16bf br = load_b(wr, 448, 32 * k, lane);
                ar = wmma_bf16(a, br, ar);
                v16bf bz = load_b(wz, 448, 32 * k, lane);
                az = wmma_bf16(a, bz, az);
                if (k < 7) {            // za half -> i_n
                    v16bf bi = load_b(wi, 224, 32 * k, lane);
                    ain = wmma_bf16(a, bi, ain);
                } else {                // h half -> h_n
                    v16bf bh = load_b(wh, 224, 32 * (k - 7), lane);
                    ahn = wmma_bf16(a, bh, ahn);
                }
            }
            int col = nt * 16 + colL;
            if (col < 200) {
                float b_r = gb_ih[col] + gb_hh[col];
                float b_z = gb_ih[200 + col] + gb_hh[200 + col];
                float b_i = gb_ih[400 + col];
                float b_h = gb_hh[400 + col];
                #pragma unroll
                for (int j = 0; j < 8; ++j) {
                    int r = j + 8 * hi;
                    float rg = sigmf(ar[j] + b_r);
                    float zg = sigmf(az[j] + b_z);
                    float ng = tanh_fast(ain[j] + b_i + rg * (ahn[j] + b_h));
                    float hn = (1.f - zg) * ng + zg * (sH[r * 200 + col] * sMask[r]);
                    sH[r * 200 + col]   = hn;
                    sApo[r * 480 + col] = (__bf16)hn;
                    feat[(rowbase + r) * 230 + col] = hn;
                    if (t == TT - 1) lasth[(size_t)(b0 + r) * 200 + col] = hn;
                }
            }
        }
        __syncthreads();

        // ---- GEMM3: post_in = elu([h_new|e] @ Wpc^T + post_h_b) -> sApi ----
        for (int nt = wid; nt < 13; nt += 8) {
            v8f acc = wmma_tile(sApo, 480, 0, 15, Wpc + (size_t)(nt * 16) * 480, 480, lane);
            int col = nt * 16 + colL;
            if (col < 200) {
                float b = poh_b[col];
                #pragma unroll
                for (int j = 0; j < 8; ++j) {
                    int r = j + 8 * hi;
                    sApi[r * 224 + col] = (__bf16)eluf(acc[j] + b);
                }
            }
        }
        __syncthreads();

        // ---- GEMM4: post = post_in @ Wpo^T + post_b ----
        for (int nt = wid; nt < 4; nt += 8) {
            v8f acc = wmma_tile(sApi, 224, 0, 7, Wpo + (size_t)(nt * 16) * 224, 224, lane);
            int col = nt * 16 + colL;
            if (col < 60) {
                float b = po_b[col];
                #pragma unroll
                for (int j = 0; j < 8; ++j) {
                    int r = j + 8 * hi;
                    float v = acc[j] + b;
                    posts[(rowbase + r) * 60 + col] = v;
                    sPost[r * 64 + col] = v;
                }
            }
        }
        __syncthreads();

        // ---- sample: z' = mean + (softplus(std_raw)+0.1)*noise ----
        for (int i = tid; i < 16 * 30; i += 256) {
            int r = i / 30, c = i - r * 30;
            float mean = sPost[r * 64 + c];
            float stdv = softplusf(sPost[r * 64 + 30 + c]) + 0.1f;
            float smp  = mean + stdv * noise[(rowbase + r) * 30 + c];
            sZ[i] = smp;
            feat[(rowbase + r) * 230 + 200 + c] = smp;
            if (t == TT - 1) lastz[(size_t)(b0 + r) * 30 + c] = smp;
        }
        __syncthreads();
    }
}

// ---------------- priors: fully parallel two-stage GEMM over all T*B rows ----------------
__global__ void __launch_bounds__(128) rssm_priors(
    const float* __restrict__ feat,
    const float* __restrict__ phb, const float* __restrict__ pb,
    const __bf16* __restrict__ Wprh, const __bf16* __restrict__ Wpr,
    float* __restrict__ priors)
{
    __shared__ __attribute__((aligned(64))) __bf16 pA[4][16 * 224];
    __shared__ __attribute__((aligned(64))) __bf16 pB[4][16 * 224];

    const int tid  = threadIdx.x;
    const int lane = tid & 31;
    const int wid  = tid >> 5;
    const int hi   = lane >> 4;
    const int colL = lane & 15;
    const size_t tile = (size_t)blockIdx.x * 4 + wid;   // 16384 tiles total
    const size_t g0 = tile * 16;

    __bf16* A  = pA[wid];
    __bf16* Bm = pB[wid];

    for (int i = lane; i < 16 * 224; i += 32) {
        int r = i / 224, k = i - r * 224;
        A[i] = (k < 200) ? (__bf16)feat[(g0 + r) * 230 + k] : (__bf16)0.f;
    }
    for (int i = lane; i < 16 * 24; i += 32) {
        int r = i / 24, k = 200 + (i - r * 24);
        Bm[r * 224 + k] = (__bf16)0.f;
    }
    __syncthreads();

    for (int nt = 0; nt < 13; ++nt) {
        v8f acc = wmma_tile(A, 224, 0, 7, Wprh + (size_t)(nt * 16) * 224, 224, lane);
        int col = nt * 16 + colL;
        if (col < 200) {
            float b = phb[col];
            #pragma unroll
            for (int j = 0; j < 8; ++j) {
                int r = j + 8 * hi;
                Bm[r * 224 + col] = (__bf16)eluf(acc[j] + b);
            }
        }
    }
    __syncthreads();

    for (int nt = 0; nt < 4; ++nt) {
        v8f acc = wmma_tile(Bm, 224, 0, 7, Wpr + (size_t)(nt * 16) * 224, 224, lane);
        int col = nt * 16 + colL;
        if (col < 60) {
            float b = pb[col];
            #pragma unroll
            for (int j = 0; j < 8; ++j) {
                int r = j + 8 * hi;
                priors[(g0 + r) * 60 + col] = acc[j] + b;
            }
        }
    }
}

// ---------------- host launcher ----------------
static inline void prep(__bf16* dst, const float* sA, int Ka, int KpA,
                        const float* sB, int Kb, int N, int Np, int Kp,
                        hipStream_t stream)
{
    int n = Np * Kp;
    prep_weight<<<(n + 255) / 256, 256, 0, stream>>>(dst, sA, Ka, KpA, sB, Kb, N, Np, Kp);
}

extern "C" void kernel_launch(void* const* d_in, const int* in_sizes, int n_in,
                              void* d_out, int out_size, void* d_ws, size_t ws_size,
                              hipStream_t stream)
{
    const float* embed  = (const float*)d_in[0];
    const float* action = (const float*)d_in[1];
    const unsigned char* reset = (const unsigned char*)d_in[2];
    const float* h0     = (const float*)d_in[3];
    const float* z0     = (const float*)d_in[4];
    /* d_in[5] glob_state unused */
    const float* noise  = (const float*)d_in[6];
    const float* z_w    = (const float*)d_in[7];
    const float* z_b    = (const float*)d_in[8];
    const float* a_w    = (const float*)d_in[9];
    const float* w_ih   = (const float*)d_in[10];
    const float* w_hh   = (const float*)d_in[11];
    const float* b_ih   = (const float*)d_in[12];
    const float* b_hh   = (const float*)d_in[13];
    const float* prh_w  = (const float*)d_in[14];
    const float* prh_b  = (const float*)d_in[15];
    const float* pr_w   = (const float*)d_in[16];
    const float* pr_b   = (const float*)d_in[17];
    const float* poh_w  = (const float*)d_in[18];
    const float* poh_b  = (const float*)d_in[19];
    const float* poe_w  = (const float*)d_in[20];
    const float* po_w   = (const float*)d_in[21];
    const float* po_b   = (const float*)d_in[22];

    __bf16* ws = (__bf16*)d_ws;

    // ---- weight prep (bf16, padded) ----
    prep(ws + O_ZA,  z_w,   30, 30,  a_w,   7,   200, 208, 64,  stream);
    prep(ws + O_RZ,  w_ih,  200, 224, w_hh, 200, 200, 208, 448, stream);                         // r gate
    prep(ws + O_RZ + (size_t)208 * 448,
                     w_ih + 200 * 200, 200, 224, w_hh + 200 * 200, 200, 200, 208, 448, stream); // z gate
    prep(ws + O_IN,  w_ih + 400 * 200, 200, 224, w_ih, 0, 200, 208, 224, stream);               // i_n
    prep(ws + O_HN,  w_hh + 400 * 200, 200, 224, w_hh, 0, 200, 208, 224, stream);               // h_n
    prep(ws + O_PC,  poh_w, 200, 224, poe_w, 256, 200, 208, 480, stream);
    prep(ws + O_PO,  po_w,  200, 224, po_w,  0,   60,  64,  224, stream);
    prep(ws + O_PRH, prh_w, 200, 224, prh_w, 0,   200, 208, 224, stream);
    prep(ws + O_PR,  pr_w,  200, 224, pr_w,  0,   60,  64,  224, stream);

    // ---- output layout: priors | posts | features | last_h | last_z ----
    float* out    = (float*)d_out;
    float* priors = out;
    float* posts  = priors + (size_t)TT * BB * 60;
    float* feat   = posts  + (size_t)TT * BB * 60;
    float* lasth  = feat   + (size_t)TT * BB * 230;
    float* lastz  = lasth  + (size_t)BB * 200;

    rssm_scan<<<BB / 16, 256, 0, stream>>>(
        embed, action, reset, h0, z0, noise,
        z_b, b_ih, b_hh, poh_b, po_b,
        ws + O_ZA, ws + O_RZ, ws + O_IN, ws + O_HN, ws + O_PC, ws + O_PO,
        posts, feat, lasth, lastz);

    rssm_priors<<<(TT * BB / 16) / 4, 128, 0, stream>>>(
        feat, prh_b, pr_b, ws + O_PRH, ws + O_PR, priors);
}
